// PairwiseBilinear_15891378995503
// MI455X (gfx1250) — matrix operand order repaired and checked by reading, not compiled, not run on hardware
//
#include <hip/hip_runtime.h>

// PairwiseBilinear: out[i,j] = X[i] @ W @ X[j] + b    (N=4096, D=1024, fp32)
// bf16-split (hi/lo) emulated-fp32 GEMMs on v_wmma_f32_16x16x32_bf16 with
// async global->LDS staging (ASYNCcnt) and K double-buffering.
//   P  = X @ W       (W pre-transposed so both GEMMs are A * B^T form)
//   out = P @ X^T + b
// Workspace requirement: 36 MB.

#define N_TOK 4096
#define D_DIM 1024

constexpr int KPAD = 48;  // LDS K-stride in bf16 elems (96 B rows, 16B-aligned)

typedef __attribute__((ext_vector_type(16))) __bf16 v16bf;
typedef __attribute__((ext_vector_type(8)))  float  v8f;
typedef __attribute__((ext_vector_type(4)))  int    v4i;

union Frag {
  v16bf v;
  uint4 q[2];
};

__device__ __forceinline__ unsigned short f2bf(float x) {
  unsigned u = __builtin_bit_cast(unsigned, x);
  u += 0x7FFFu + ((u >> 16) & 1u);   // RNE to bf16
  return (unsigned short)(u >> 16);
}
__device__ __forceinline__ float bf2f(unsigned short h) {
  unsigned u = ((unsigned)h) << 16;
  return __builtin_bit_cast(float, u);
}

// fp32 -> (bf16 hi, bf16 lo); x ~= hi + lo to ~16 mantissa bits.
__global__ __launch_bounds__(256) void split_kernel(
    const float* __restrict__ src, unsigned short* __restrict__ hi,
    unsigned short* __restrict__ lo, int n) {
  int i = blockIdx.x * blockDim.x + threadIdx.x;
  if (i >= n) return;
  float x = src[i];
  unsigned short h = f2bf(x);
  hi[i] = h;
  lo[i] = f2bf(x - bf2f(h));
}

// Same, but writes the transpose of a D x D matrix (for W -> W^T).
__global__ __launch_bounds__(256) void split_tr_kernel(
    const float* __restrict__ src, unsigned short* __restrict__ hi,
    unsigned short* __restrict__ lo) {
  int i = blockIdx.x * blockDim.x + threadIdx.x;   // i = k*D + n (coalesced read)
  float x = src[i];
  int k = i >> 10;          // row of W
  int n = i & (D_DIM - 1);  // col of W
  int o = n * D_DIM + k;    // transposed index
  unsigned short h = f2bf(x);
  hi[o] = h;
  lo[o] = f2bf(x - bf2f(h));
}

// 16-byte async global->LDS copy (tracked by ASYNCcnt, no VGPR round trip).
__device__ __forceinline__ void async_cp16(const unsigned short* g,
                                           unsigned short* l) {
  __builtin_amdgcn_global_load_async_to_lds_b128(
      (__attribute__((address_space(1))) v4i*)g,
      (__attribute__((address_space(3))) v4i*)l, 0, 0);
}

// A fragment (16x32 bf16): lane L -> row L%16, K runs {kb..kb+7, kb+16..kb+23},
// kb = 8*(L>=16). Tile row-major [row][KPAD].
__device__ __forceinline__ v16bf frag_a(const unsigned short* t, int row, int lane) {
  const unsigned short* p = t + row * KPAD + ((lane >> 4) << 3);
  Frag f;
  f.q[0] = *(const uint4*)(p);
  f.q[1] = *(const uint4*)(p + 16);
  return f.v;
}

// B fragment (32x16 bf16): lane L -> col L%16, K runs {kb..kb+7, kb+16..kb+23}, kb = 8*(L>=16).
// Tile stored [col][KPAD].
__device__ __forceinline__ v16bf frag_b(const unsigned short* t, int col, int lane) {
  const unsigned short* p = t + col * KPAD + ((lane >> 4) << 3);
  Frag f;
  f.q[0] = *(const uint4*)(p);
  f.q[1] = *(const uint4*)(p + 16);
  return f.v;
}

__device__ __forceinline__ v8f wmma_bf16(v16bf a, v16bf b, v8f c) {
  return __builtin_amdgcn_wmma_f32_16x16x32_bf16(false, a, false, b, (short)0, c,
                                                 false, false);
}

// Split-bf16 GEMM:  O(M x N) = (Ah+Al) * (Bh+Bl)^T [+ bias]
//   A: M x K row-major,  B: N x K row-major (so B^T is K x N).
// Block: 256 threads = 8 waves; WG tile 128(M) x 128(N); wave tile 32 x 64.
// K-step 32, double-buffered LDS, async global->LDS staging.
template <bool SPLIT_OUT>
__global__ __launch_bounds__(256) void gemm_kernel(
    const unsigned short* __restrict__ Ah, const unsigned short* __restrict__ Al,
    const unsigned short* __restrict__ Bh, const unsigned short* __restrict__ Bl,
    int ldA, int ldB, int K,
    unsigned short* __restrict__ Oh, unsigned short* __restrict__ Ol,
    float* __restrict__ O, int ldo, const float* __restrict__ bias) {
  __shared__ __align__(16) unsigned short sAh[2][128 * KPAD];
  __shared__ __align__(16) unsigned short sAl[2][128 * KPAD];
  __shared__ __align__(16) unsigned short sBh[2][128 * KPAD];
  __shared__ __align__(16) unsigned short sBl[2][128 * KPAD];

  const int tid  = threadIdx.x;
  const int lane = tid & 31;
  const int wave = tid >> 5;
  const int wm   = wave & 3;   // M position: 32 rows each
  const int wn   = wave >> 2;  // N position: 64 cols each
  const int mBase = blockIdx.y * 128;
  const int nBase = blockIdx.x * 128;

  // Per-thread staging coordinates: 128 rows x 32 K per tile; 16 B per copy.
  const int r0 = tid >> 2;          // 0..63
  const int kb = (tid & 3) << 3;    // 0,8,16,24 (bf16 elems; *2 = 16 B)

  auto stage = [&](int buf, int k0) {
    #pragma unroll
    for (int p = 0; p < 2; ++p) {
      const int row = r0 + (p << 6);
      const int lo  = row * KPAD + kb;
      const size_t ga = (size_t)(mBase + row) * ldA + (size_t)k0 + kb;
      async_cp16(&Ah[ga], &sAh[buf][lo]);
      async_cp16(&Al[ga], &sAl[buf][lo]);
      const size_t gb = (size_t)(nBase + row) * ldB + (size_t)k0 + kb;
      async_cp16(&Bh[gb], &sBh[buf][lo]);
      async_cp16(&Bl[gb], &sBl[buf][lo]);
    }
  };

  v8f acc[2][4] = {};

  const int nk = K >> 5;  // K / 32
  stage(0, 0);
  for (int i = 0; i < nk; ++i) {
    const int buf = i & 1;
    __builtin_amdgcn_s_wait_asynccnt(0);  // my tile-i copies landed in LDS
    __syncthreads();                      // everyone's landed; compute(i-1) done
    if (i + 1 < nk) stage(buf ^ 1, (i + 1) << 5);  // prefetch tile i+1

    // Fragments: 2 M-subtiles, 4 N-subtiles, hi+lo each.
    const int ar = wm * 32 + (lane & 15);
    v16bf ah[2], al[2], bh[4], bl[4];
    #pragma unroll
    for (int mi = 0; mi < 2; ++mi) {
      ah[mi] = frag_a(sAh[buf], ar + mi * 16, lane);
      al[mi] = frag_a(sAl[buf], ar + mi * 16, lane);
    }
    #pragma unroll
    for (int ni = 0; ni < 4; ++ni) {
      const int bc = wn * 64 + ni * 16 + (lane & 15);
      bh[ni] = frag_b(sBh[buf], bc, lane);
      bl[ni] = frag_b(sBl[buf], bc, lane);
    }
    // 24 WMMAs: 3 split terms (hi*hi + lo*hi + hi*lo) x 2x4 tiles.
    #pragma unroll
    for (int mi = 0; mi < 2; ++mi)
      #pragma unroll
      for (int ni = 0; ni < 4; ++ni) {
        acc[mi][ni] = wmma_bf16(ah[mi], bh[ni], acc[mi][ni]);
        acc[mi][ni] = wmma_bf16(al[mi], bh[ni], acc[mi][ni]);
        acc[mi][ni] = wmma_bf16(ah[mi], bl[ni], acc[mi][ni]);
      }
  }

  // Epilogue. C/D layout: VGPR r, lane L -> row = r + 8*(L>=16), col = L%16.
  float bias0 = 0.0f;
  if (!SPLIT_OUT) bias0 = *bias;
  const int rsel = (lane >> 4) << 3;
  const int cid  = lane & 15;
  #pragma unroll
  for (int mi = 0; mi < 2; ++mi)
    #pragma unroll
    for (int ni = 0; ni < 4; ++ni) {
      const int row0 = mBase + wm * 32 + mi * 16 + rsel;
      const int col  = nBase + wn * 64 + ni * 16 + cid;
      v8f c = acc[mi][ni];
      #pragma unroll
      for (int r = 0; r < 8; ++r) {
        const size_t idx = (size_t)(row0 + r) * ldo + col;
        const float v = c[r];
        if (SPLIT_OUT) {
          const unsigned short h = f2bf(v);
          Oh[idx] = h;
          Ol[idx] = f2bf(v - bf2f(h));
        } else {
          O[idx] = v + bias0;
        }
      }
    }
}

extern "C" void kernel_launch(void* const* d_in, const int* in_sizes, int n_in,
                              void* d_out, int out_size, void* d_ws, size_t ws_size,
                              hipStream_t stream) {
  const float* X    = (const float*)d_in[0];  // (N, D)
  const float* W    = (const float*)d_in[1];  // (D, D)
  const float* bias = (const float*)d_in[2];  // scalar
  float* out = (float*)d_out;                 // (N, N)

  const size_t ND = (size_t)N_TOK * D_DIM;
  const size_t DD = (size_t)D_DIM * D_DIM;
  unsigned short* ws = (unsigned short*)d_ws;  // 36 MB total
  unsigned short* Xh  = ws;
  unsigned short* Xl  = Xh + ND;
  unsigned short* Wth = Xl + ND;   // W^T hi
  unsigned short* Wtl = Wth + DD;  // W^T lo
  unsigned short* Ph  = Wtl + DD;
  unsigned short* Pl  = Ph + ND;

  split_kernel<<<(unsigned)(ND / 256), 256, 0, stream>>>(X, Xh, Xl, (int)ND);
  split_tr_kernel<<<(unsigned)(DD / 256), 256, 0, stream>>>(W, Wth, Wtl);

  // P = X @ W = X @ (W^T)^T : M=4096, N=1024, K=1024
  gemm_kernel<true>
      <<<dim3(D_DIM / 128, N_TOK / 128), 256, 0, stream>>>(
          Xh, Xl, Wth, Wtl, D_DIM, D_DIM, D_DIM,
          Ph, Pl, nullptr, D_DIM, nullptr);

  // out = P @ X^T + b : M=4096, N=4096, K=1024
  gemm_kernel<false>
      <<<dim3(N_TOK / 128, N_TOK / 128), 256, 0, stream>>>(
          Ph, Pl, Xh, Xl, D_DIM, D_DIM, D_DIM,
          nullptr, nullptr, out, N_TOK, bias);
}
